// SSIMLoss_3848290697568
// MI455X (gfx1250) — matrix operand order repaired and checked by reading, not compiled, not run on hardware
//
#include <hip/hip_runtime.h>

typedef __attribute__((ext_vector_type(16))) _Float16 v16h;
typedef __attribute__((ext_vector_type(8)))  _Float16 v8h;
typedef __attribute__((ext_vector_type(8)))  float    v8f;

#if defined(__has_builtin)
#if __has_builtin(__builtin_amdgcn_wmma_f16_16x16x32_f16)
#define HAVE_WMMA_F16_ACC 1
#endif
#endif
#ifndef HAVE_WMMA_F16_ACC
#define HAVE_WMMA_F16_ACC 0
#endif

#define IMG_H 512
#define IMG_W 512
#define NBC   96                              // 32 batch * 3 channels
#define WAVES_PER_BLOCK 8                     // block covers 128x64 outputs
#define GRID_X 4                              // 4 * 128 = 512 cols
#define STRIPS_Y 8                            // 512/64 rows
#define TILES_PER_STRIP 4                     // 4 x 16-row tiles per strip
#define NBLOCKS (GRID_X * STRIPS_Y * NBC)     // 3072

#define LDS_STRIDE 156                        // f32/row (148 + pad; 156%64=28 -> conflict-free b128)
#define NGROUPS 37                            // float4 groups/row (148 cols: 128 + halo, 16B aligned)

// Banded Gaussian weight operand, B layout (32x16): elem j of lane L holds
// W[K=k][N=L&15] = g[k-(L&15)], k=ko+j+(j>=8?8:0), ko=(L<16)?0:8. Serves BOTH the
// horizontal pass and the transposed vertical pass.
struct BTab {
  float w[32][16];
  constexpr BTab() : w{} {
    constexpr float G[11] = {
        0.00102837f, 0.00759876f, 0.03600071f, 0.10936070f, 0.21300553f,
        0.26601210f,
        0.21300553f, 0.10936070f, 0.03600071f, 0.00759876f, 0.00102837f};
    for (int l = 0; l < 32; ++l) {
      const int nn  = l & 15;
      const int kko = (l < 16) ? 0 : 8;
      for (int j = 0; j < 16; ++j) {
        const int i = (kko + j + (j >= 8 ? 8 : 0)) - nn;
        w[l][j] = (i >= 0 && i < 11) ? G[i] : 0.0f;
      }
    }
  }
};
__constant__ BTab g_btab;

__global__ __launch_bounds__(256)
void ssim_wmma_strips(const float* __restrict__ img1,
                      const float* __restrict__ img2,
                      float* __restrict__ blockSums) {
  const int lane = threadIdx.x & 31;
  const int wave = threadIdx.x >> 5;
  const int row_base = blockIdx.y * 64;            // strip top row
  const int colBase  = (int)blockIdx.x * 128 - 8;  // 16B-aligned staging start
  const size_t planeOff = (size_t)blockIdx.z * IMG_H * IMG_W;
  const float* __restrict__ p1 = img1 + planeOff;
  const float* __restrict__ p2 = img2 + planeOff;

  const int n     = lane & 15;
  const int base0 = (lane < 16) ? 0 : 8;     // A-operand K grouping per half

  // Double-buffered f32 chunk staging: [buffer][image][16 x LDS_STRIDE]
  __shared__ float sbuf[2][2][16 * LDS_STRIDE];    // 39.9 KB

  // Shared banded-Gaussian operand for both passes.
  v16h Bw;
  {
    const float* __restrict__ wr = g_btab.w[lane];
#pragma unroll
    for (int j = 0; j < 16; ++j) Bw[j] = (_Float16)wr[j];
  }

  // ---- cooperative, coalesced, division-free staging ---------------------
  // 32-bit element indices -> saddr + voffset addressing; group validity is
  // uniform (groups 4-aligned, width % 4 == 0): one predicate per group.
  auto stage = [&](int c, int b) {
    const int rr = (int)threadIdx.x >> 4;          // 0..15 row within chunk
    const int gc = (int)threadIdx.x & 15;          // base float4 group
    const int r  = row_base - 5 + 16 * c + rr;
    const bool rOK = (unsigned)r < (unsigned)IMG_H;
    const int rowOff = (rOK ? r : 0) << 9;         // * IMG_W, 32-bit
    float* d1 = &sbuf[b][0][rr * LDS_STRIDE];
    float* d2 = &sbuf[b][1][rr * LDS_STRIDE];
#pragma unroll
    for (int i = 0; i < 3; ++i) {
      const int g = gc + 16 * i;                   // i<2 always in range
      if (g < NGROUPS) {
        const int  cb  = colBase + 4 * g;          // 16B aligned global col
        const bool gOK = rOK && ((unsigned)cb <= (unsigned)(IMG_W - 4));
        const int  cbSafe = (cb < 0) ? 0 : ((cb > IMG_W - 4) ? (IMG_W - 4) : cb);
        const int  off = rowOff + cbSafe;          // 32-bit index
        const float4 f1 = *reinterpret_cast<const float4*>(p1 + off);
        const float4 f2 = *reinterpret_cast<const float4*>(p2 + off);
        const float* e1 = reinterpret_cast<const float*>(&f1);
        const float* e2 = reinterpret_cast<const float*>(&f2);
        const int abase = 4 * g - 3;               // block-independent A index
#pragma unroll
        for (int e = 0; e < 4; ++e) {
          const float v1 = gOK ? e1[e] : 0.0f;     // zero "SAME" padding
          const float v2 = gOK ? e2[e] : 0.0f;
          if (abase + e >= 0) { d1[abase + e] = v1; d2[abase + e] = v2; }
        }
      }
    }
  };

  // ---- gather this wave's A-operands from LDS (4 aligned b128 / image) ---
  auto consume = [&](int b, v16h& axc, v16h& ayc) {
    const int o = n * LDS_STRIDE + 16 * wave + base0;  // 16B aligned
    const float* r1 = &sbuf[b][0][o];
    const float* r2 = &sbuf[b][1][o];
    float t1[16], t2[16];
    *reinterpret_cast<float4*>(&t1[0])  = *reinterpret_cast<const float4*>(r1 + 0);
    *reinterpret_cast<float4*>(&t1[4])  = *reinterpret_cast<const float4*>(r1 + 4);
    *reinterpret_cast<float4*>(&t1[8])  = *reinterpret_cast<const float4*>(r1 + 16);
    *reinterpret_cast<float4*>(&t1[12]) = *reinterpret_cast<const float4*>(r1 + 20);
    *reinterpret_cast<float4*>(&t2[0])  = *reinterpret_cast<const float4*>(r2 + 0);
    *reinterpret_cast<float4*>(&t2[4])  = *reinterpret_cast<const float4*>(r2 + 4);
    *reinterpret_cast<float4*>(&t2[8])  = *reinterpret_cast<const float4*>(r2 + 16);
    *reinterpret_cast<float4*>(&t2[12]) = *reinterpret_cast<const float4*>(r2 + 20);
#pragma unroll
    for (int j = 0; j < 16; ++j) {
      axc[j] = (_Float16)t1[j];                    // packs into v_cvt_pk
      ayc[j] = (_Float16)t2[j];
    }
  };

#if HAVE_WMMA_F16_ACC
  // H pass accumulates in f16: D layout == V-pass A-operand halves, so the
  // vertical A operand is a pure register concat (no converts at all).
  v8h zeroh;
#pragma unroll
  for (int j = 0; j < 8; ++j) zeroh[j] = (_Float16)0.0f;
  v8h Hprev[5], Hcur[5];
  auto hpass = [&](const v16h& axc, const v16h& ayc, v8h* H) {
    H[0] = __builtin_amdgcn_wmma_f16_16x16x32_f16(
        false, axc, false, Bw, (short)0, zeroh, false, false);
    H[1] = __builtin_amdgcn_wmma_f16_16x16x32_f16(
        false, ayc, false, Bw, (short)0, zeroh, false, false);
    H[2] = __builtin_amdgcn_wmma_f16_16x16x32_f16(
        false, axc * axc, false, Bw, (short)0, zeroh, false, false);
    H[3] = __builtin_amdgcn_wmma_f16_16x16x32_f16(
        false, ayc * ayc, false, Bw, (short)0, zeroh, false, false);
    H[4] = __builtin_amdgcn_wmma_f16_16x16x32_f16(
        false, axc * ayc, false, Bw, (short)0, zeroh, false, false);
  };
#else
  const v8f zero = 0.0f;
  v8f Hprev[5], Hcur[5];
  auto hpass = [&](const v16h& axc, const v16h& ayc, v8f* H) {
    H[0] = __builtin_amdgcn_wmma_f32_16x16x32_f16(
        false, axc, false, Bw, (short)0, zero, false, false);
    H[1] = __builtin_amdgcn_wmma_f32_16x16x32_f16(
        false, ayc, false, Bw, (short)0, zero, false, false);
    H[2] = __builtin_amdgcn_wmma_f32_16x16x32_f16(
        false, axc * axc, false, Bw, (short)0, zero, false, false);
    H[3] = __builtin_amdgcn_wmma_f32_16x16x32_f16(
        false, ayc * ayc, false, Bw, (short)0, zero, false, false);
    H[4] = __builtin_amdgcn_wmma_f32_16x16x32_f16(
        false, axc * ayc, false, Bw, (short)0, zero, false, false);
  };
#endif

  // ---- pipeline: chunk c -> buf[c&1]; tile t uses chunks t, t+1 ----------
  stage(0, 0);
  stage(1, 1);
  __syncthreads();

  {
    v16h ax, ay;
    consume(0, ax, ay);
    hpass(ax, ay, Hprev);
  }

  const float C1 = 1e-4f, C2 = 9e-4f;
  const v8f zeroc = 0.0f;
  float partial = 0.0f;

#pragma unroll
  for (int t = 0; t < TILES_PER_STRIP; ++t) {
    if (t < TILES_PER_STRIP - 1) stage(t + 2, t & 1);  // overlap with WMMAs
    {
      v16h ax, ay;
      consume((t + 1) & 1, ax, ay);
      hpass(ax, ay, Hcur);
    }

    // Vertical pass, transposed: out_v^T = H^T x W. H-pass D layout IS the
    // A-operand layout of H^T: elem j = (j<8) ? Hprev[j] : Hcur[j-8] for
    // both half-waves -- no shuffles, no selects.
    v8f res[5];
#pragma unroll
    for (int s = 0; s < 5; ++s) {
      v16h A2;
#if HAVE_WMMA_F16_ACC
#pragma unroll
      for (int j = 0; j < 8; ++j) {                // register concat only
        A2[j]     = Hprev[s][j];
        A2[j + 8] = Hcur[s][j];
      }
#else
#pragma unroll
      for (int j = 0; j < 8; ++j) {
        A2[j]     = (_Float16)Hprev[s][j];
        A2[j + 8] = (_Float16)Hcur[s][j];
      }
#endif
      res[s] = __builtin_amdgcn_wmma_f32_16x16x32_f16(
          false, A2, false, Bw, (short)0, zeroc, false, false);
    }

    // Pointwise SSIM on the (transposed) 16x16 map; only the sum matters.
#pragma unroll
    for (int v = 0; v < 8; ++v) {
      const float mu1 = res[0][v], mu2 = res[1][v];
      const float b11 = res[2][v], b22 = res[3][v], b12 = res[4][v];
      const float mu1s = mu1 * mu1, mu2s = mu2 * mu2, mu12 = mu1 * mu2;
      const float s1 = b11 - mu1s, s2 = b22 - mu2s, s12 = b12 - mu12;
      const float num = (2.0f * mu12 + C1) * (2.0f * s12 + C2);
      const float den = (mu1s + mu2s + C1) * (s1 + s2 + C2);
      partial += num * __builtin_amdgcn_rcpf(den);  // fast rcp, 2-ulp ok
    }

#pragma unroll
    for (int s = 0; s < 5; ++s) Hprev[s] = Hcur[s];   // register rotate
    __syncthreads();   // buf[(t+1)&1] reads done + stage(t+2) visible
  }

  // ---- wave + block reduction (deterministic, no atomics) ----------------
#pragma unroll
  for (int off = 16; off >= 1; off >>= 1)
    partial += __shfl_xor(partial, off, 32);

  __shared__ float smem[WAVES_PER_BLOCK];
  if (lane == 0) smem[wave] = partial;
  __syncthreads();
  if (threadIdx.x == 0) {
    float t = 0.0f;
#pragma unroll
    for (int w = 0; w < WAVES_PER_BLOCK; ++w) t += smem[w];
    blockSums[((size_t)blockIdx.z * gridDim.y + blockIdx.y) * gridDim.x +
              blockIdx.x] = t;   // overwrite each call: graph-replay safe
  }
}

__global__ __launch_bounds__(256)
void ssim_finalize(const float* __restrict__ blockSums, float* __restrict__ out,
                   int nblocks, float invN) {
  __shared__ float sm[256];
  float t = 0.0f;
  for (int i = threadIdx.x; i < nblocks; i += 256) t += blockSums[i];
  sm[threadIdx.x] = t;
  __syncthreads();
  for (int s = 128; s > 0; s >>= 1) {
    if (threadIdx.x < s) sm[threadIdx.x] += sm[threadIdx.x + s];
    __syncthreads();
  }
  if (threadIdx.x == 0) out[0] = 1.0f - sm[0] * invN;
}

extern "C" void kernel_launch(void* const* d_in, const int* in_sizes, int n_in,
                              void* d_out, int out_size, void* d_ws,
                              size_t ws_size, hipStream_t stream) {
  (void)in_sizes; (void)n_in; (void)out_size; (void)ws_size;
  const float* img1 = (const float*)d_in[0];
  const float* img2 = (const float*)d_in[1];
  float* blockSums  = (float*)d_ws;     // NBLOCKS floats of scratch
  float* out        = (float*)d_out;

  dim3 grid(GRID_X, STRIPS_Y, NBC);     // 4 x 8 x 96 blocks, 8 waves each
  dim3 block(256);
  ssim_wmma_strips<<<grid, block, 0, stream>>>(img1, img2, blockSums);

  const float invN = 1.0f / (32.0f * 3.0f * 512.0f * 512.0f);
  ssim_finalize<<<1, 256, 0, stream>>>(blockSums, out, NBLOCKS, invN);
}
